// GaussianRenderer_20289425506732
// MI455X (gfx1250) — matrix operand order repaired and checked by reading, not compiled, not run on hardware
//
#include <hip/hip_runtime.h>
#include <hip/hip_bf16.h>

// ---------------- problem constants ----------------
#define TDIM 768      // TRANSFORMER_DIM (K of GEMM1)
#define ODIM 448      // OUT_DIM (N of GEMM1, K and N of GEMM2)
#define BM   64       // tokens per block
#define NT   28       // ODIM / 16
#define KT1  24       // TDIM / 32
#define KT2  14       // ODIM / 32
#define AF_STRIDE 36  // padded LDS stride (fp32) for A tile (bank-conflict free)
#define AF_TILE   (BM * AF_STRIDE)          // floats per A buffer
#define AF_BYTES  (AF_TILE * 4)             // bytes per A buffer (9216)
#define H_STRIDE 456  // padded LDS stride (bf16) for h / f tiles
#define THREADS 512   // 16 waves
#define ABUFS 4       // async A-tile pipeline depth

typedef __attribute__((ext_vector_type(16))) __bf16 v16bf;
typedef __attribute__((ext_vector_type(8)))  __bf16 v8bf;
typedef __attribute__((ext_vector_type(8)))  float  v8f;

union V16U { v16bf v; v8bf h[2]; };

__device__ __forceinline__ float fast_sigmoid(float x) {
    return 1.0f / (1.0f + __expf(-x));
}

// Async copy 16 bytes/lane from global to LDS (GLOBAL_LOAD_ASYNC_TO_LDS_B128,
// tracked by ASYNCcnt). dst_lds = LDS byte offset, src = global address.
__device__ __forceinline__ void async_copy_b128(uint32_t dst_lds, const void* src) {
    asm volatile("global_load_async_to_lds_b128 %0, %1, off"
                 :: "v"(dst_lds), "v"((uint64_t)(uintptr_t)src)
                 : "memory");
}
// ASYNCcnt completes in-order for async loads: waiting <=N with N+1 in flight
// guarantees the oldest outstanding copy has fully landed in LDS.
__device__ __forceinline__ void wait_async2() {
    asm volatile("s_wait_asynccnt 0x2" ::: "memory");
}
__device__ __forceinline__ void wait_async1() {
    asm volatile("s_wait_asynccnt 0x1" ::: "memory");
}
__device__ __forceinline__ void wait_async0() {
    asm volatile("s_wait_asynccnt 0x0" ::: "memory");
}

// -----------------------------------------------------------------------------
// Pack W (K x N, row-major fp32) into bf16 WMMA B-fragment order:
//   fragment block (kt, nt) = 512 bf16, laid out [lane 0..31][elem 0..15]
//   lane l: n = nt*16 + (l & 15), elem e: k = kt*32 + (l>>4)*8 + (e & 7) + (e>>3)*16
// A wave then loads one whole fragment as lane-contiguous 32B (2 x b128/lane).
// -----------------------------------------------------------------------------
__global__ void pack_weights(const float* __restrict__ W1,
                             const float* __restrict__ W2,
                             __bf16* __restrict__ pW1,
                             __bf16* __restrict__ pW2) {
    int idx = blockIdx.x * blockDim.x + threadIdx.x;
    const int P1 = TDIM * ODIM;
    const int P2 = ODIM * ODIM;
    const float* W;
    __bf16* P;
    if (idx < P1)             { W = W1; P = pW1; }
    else if (idx < P1 + P2)   { idx -= P1; W = W2; P = pW2; }
    else return;
    int e   = idx & 15;
    int l   = (idx >> 4) & 31;
    int blk = idx >> 9;
    int nt  = blk % NT;
    int kt  = blk / NT;
    int n   = nt * 16 + (l & 15);
    int k   = kt * 32 + (l >> 4) * 8 + (e & 7) + (e >> 3) * 16;
    P[idx] = (__bf16)W[k * ODIM + n];
}

// -----------------------------------------------------------------------------
// Fused MLP + gaussian epilogue.
//   Block: 64 tokens, 512 threads = 16 waves.
//   Wave w: M-tile mt = w>>2 (16 rows), N-tiles qh*7 .. qh*7+6 (qh = w&3).
// -----------------------------------------------------------------------------
__global__ __launch_bounds__(THREADS) void mlp_gs_kernel(
    const float* __restrict__ xyzs,
    const float* __restrict__ latents,
    const float* __restrict__ b1,
    const float* __restrict__ b2,
    const __bf16* __restrict__ pW1,
    const __bf16* __restrict__ pW2,
    float* __restrict__ out)
{
    __shared__ float  Afp[ABUFS][AF_TILE];   // latents tiles, fp32, 4-deep
    __shared__ __bf16 hbuf[BM * H_STRIDE];   // silu(x W1 + b1), bf16
    __shared__ __bf16 fbuf[BM * H_STRIDE];   // h W2 + b2, bf16

    const int tid  = threadIdx.x;
    const int lane = tid & 31;
    const int wave = tid >> 5;
    const int mt   = wave >> 2;          // 0..3
    const int qh   = wave & 3;           // 0..3  -> ntiles qh*7..qh*7+6
    const int r    = lane & 15;
    const int hb   = lane >> 4;
    const int base = blockIdx.x * BM;    // first token of this block

    // async staging coords: each thread moves one float4 of the 64x32 fp32 tile
    const int srow = tid >> 3;           // 0..63
    const int scol = (tid & 7) * 4;      // 0..28
    const float* lat_row = latents + (size_t)(base + srow) * TDIM + scol;
    const uint32_t lds_dst0 =
        (uint32_t)(uintptr_t)&Afp[0][srow * AF_STRIDE + scol];
    const float* abase = &Afp[0][(mt * 16 + r) * AF_STRIDE + hb * 8];

    v8f acc[7];
    #pragma unroll
    for (int j = 0; j < 7; ++j)
        #pragma unroll
        for (int v = 0; v < 8; ++v) acc[j][v] = 0.0f;

    // one GEMM1 K-step: A frag from LDS buffer (kt&3), 7 B frags, 7 WMMAs
    auto gemm1_step = [&](int kt) {
        const float* ap = abase + (kt & (ABUFS - 1)) * AF_TILE;
        float4 a0 = *(const float4*)(ap);
        float4 a1 = *(const float4*)(ap + 4);
        float4 a2 = *(const float4*)(ap + 16);
        float4 a3 = *(const float4*)(ap + 20);
        v16bf av;
        av[0]  = (__bf16)a0.x; av[1]  = (__bf16)a0.y;
        av[2]  = (__bf16)a0.z; av[3]  = (__bf16)a0.w;
        av[4]  = (__bf16)a1.x; av[5]  = (__bf16)a1.y;
        av[6]  = (__bf16)a1.z; av[7]  = (__bf16)a1.w;
        av[8]  = (__bf16)a2.x; av[9]  = (__bf16)a2.y;
        av[10] = (__bf16)a2.z; av[11] = (__bf16)a2.w;
        av[12] = (__bf16)a3.x; av[13] = (__bf16)a3.y;
        av[14] = (__bf16)a3.z; av[15] = (__bf16)a3.w;

        const __bf16* bp = pW1 + ((size_t)(kt * NT + qh * 7) << 9) + lane * 16;
        v16bf bf[7];
        #pragma unroll
        for (int j = 0; j < 7; ++j)
            bf[j] = *(const v16bf*)(bp + j * 512);
        #pragma unroll
        for (int j = 0; j < 7; ++j)
            acc[j] = __builtin_amdgcn_wmma_f32_16x16x32_bf16(
                false, av, false, bf[j], (short)0, acc[j], false, false);
    };

    // prologue: put tiles 0..2 in flight, ensure tile 0 has landed
    #pragma unroll
    for (int p = 0; p < ABUFS - 1; ++p)
        async_copy_b128(lds_dst0 + p * AF_BYTES, lat_row + p * 32);
    wait_async2();
    __syncthreads();

    // ---------------- GEMM1: h = latents(64x768) * W1(768x448) ----------------
    // main loop: branch-free body, 3 async tiles always in flight
    for (int kt = 0; kt < KT1 - (ABUFS - 1); ++kt) {
        async_copy_b128(lds_dst0 + ((kt + ABUFS - 1) & (ABUFS - 1)) * AF_BYTES,
                        lat_row + (kt + ABUFS - 1) * 32);
        gemm1_step(kt);
        wait_async2();      // tile kt+1 resident (in-order completion)
        __syncthreads();
    }
    // peeled tail: drain the async pipeline without branches in the bodies
    gemm1_step(KT1 - 3); wait_async1(); __syncthreads();
    gemm1_step(KT1 - 2); wait_async0(); __syncthreads();
    gemm1_step(KT1 - 1);

    // epilogue 1: bias + SiLU, write bf16 h tile (C layout: n=lane%16, M=v+8*hb)
    #pragma unroll
    for (int j = 0; j < 7; ++j) {
        int n = (qh * 7 + j) * 16 + r;
        float bias = b1[n];
        #pragma unroll
        for (int v = 0; v < 8; ++v) {
            float x = acc[j][v] + bias;
            hbuf[(mt * 16 + v + 8 * hb) * H_STRIDE + n] = (__bf16)(x * fast_sigmoid(x));
            acc[j][v] = 0.0f;
        }
    }
    __syncthreads();

    // ---------------- GEMM2: f = h(64x448) * W2(448x448) ----------------
    for (int kt = 0; kt < KT2; ++kt) {
        const __bf16* ap = &hbuf[(mt * 16 + r) * H_STRIDE + kt * 32 + hb * 8];
        V16U a;
        a.h[0] = *(const v8bf*)(ap);
        a.h[1] = *(const v8bf*)(ap + 16);

        const __bf16* bp = pW2 + ((size_t)(kt * NT + qh * 7) << 9) + lane * 16;
        v16bf bf[7];
        #pragma unroll
        for (int j = 0; j < 7; ++j)
            bf[j] = *(const v16bf*)(bp + j * 512);
        #pragma unroll
        for (int j = 0; j < 7; ++j)
            acc[j] = __builtin_amdgcn_wmma_f32_16x16x32_bf16(
                false, a.v, false, bf[j], (short)0, acc[j], false, false);
    }

    // epilogue 2: bias, park f in LDS for the cross-column gaussian math
    #pragma unroll
    for (int j = 0; j < 7; ++j) {
        int n = (qh * 7 + j) * 16 + r;
        float bias = b2[n];
        #pragma unroll
        for (int v = 0; v < 8; ++v)
            fbuf[(mt * 16 + v + 8 * hb) * H_STRIDE + n] = (__bf16)(acc[j][v] + bias);
    }
    __syncthreads();

    // ---------------- gaussian epilogue: 2048 gaussians / block ----------------
    #pragma unroll
    for (int g = 0; g < 4; ++g) {
        int gi   = tid + THREADS * g;    // 0..2047
        int tl   = gi >> 5;              // local token
        int slot = gi & 31;              // gaussian within token

        float f[14];
        const __bf16* fp = &fbuf[tl * H_STRIDE + slot * 14];
        #pragma unroll
        for (int d = 0; d < 14; ++d) f[d] = (float)fp[d];

        int token = base + tl;
        float x0 = xyzs[token * 3 + 0];
        float x1 = xyzs[token * 3 + 1];
        float x2 = xyzs[token * 3 + 2];

        float o[14];
        o[0] = x0 + fast_sigmoid(f[0]) * 0.2f;
        o[1] = x1 + fast_sigmoid(f[1]) * 0.2f;
        o[2] = x2 + fast_sigmoid(f[2]) * 0.2f;
        o[3] = fast_sigmoid(f[3]);
        o[4] = fast_sigmoid(f[4]) * 0.02f;
        o[5] = fast_sigmoid(f[5]) * 0.02f;
        o[6] = fast_sigmoid(f[6]) * 0.02f;
        float nrm = sqrtf(f[7]*f[7] + f[8]*f[8] + f[9]*f[9] + f[10]*f[10]);
        float inv = 1.0f / fmaxf(nrm, 1e-12f);
        o[7]  = f[7]  * inv;
        o[8]  = f[8]  * inv;
        o[9]  = f[9]  * inv;
        o[10] = f[10] * inv;
        o[11] = fminf(fmaxf(f[11], -0.5f), 0.5f) + 0.5f;
        o[12] = fminf(fmaxf(f[12], -0.5f), 0.5f) + 0.5f;
        o[13] = fminf(fmaxf(f[13], -0.5f), 0.5f) + 0.5f;

        float* op = out + ((size_t)token * 32 + slot) * 14;
        #pragma unroll
        for (int d = 0; d < 7; ++d)
            *(float2*)(op + d * 2) = make_float2(o[2 * d], o[2 * d + 1]);
    }
}

// -----------------------------------------------------------------------------
extern "C" void kernel_launch(void* const* d_in, const int* in_sizes, int n_in,
                              void* d_out, int out_size, void* d_ws, size_t ws_size,
                              hipStream_t stream) {
    const float* xyzs    = (const float*)d_in[0];
    const float* latents = (const float*)d_in[1];
    const float* W1      = (const float*)d_in[2];
    const float* b1      = (const float*)d_in[3];
    const float* W2      = (const float*)d_in[4];
    const float* b2      = (const float*)d_in[5];
    float* out = (float*)d_out;

    __bf16* pW1 = (__bf16*)d_ws;
    __bf16* pW2 = (__bf16*)((char*)d_ws + (size_t)TDIM * ODIM * sizeof(__bf16));

    const int packElems = TDIM * ODIM + ODIM * ODIM;
    pack_weights<<<(packElems + 255) / 256, 256, 0, stream>>>(W1, W2, pW1, pW2);

    const int tokens = in_sizes[1] / TDIM;   // 8 * 8192 = 65536
    mlp_gs_kernel<<<tokens / BM, THREADS, 0, stream>>>(
        xyzs, latents, b1, b2, pW1, pW2, out);
}
